// HL_LH_block_1_11484742549816
// MI455X (gfx1250) — matrix operand order repaired and checked by reading, not compiled, not run on hardware
//
#include <hip/hip_runtime.h>

// ---------------------------------------------------------------------------
// CDNA5 (gfx1250) implementation of the EGAT + Path_Complex forward pass.
// Dense GEMMs run on bf16 WMMA (v_wmma_f32_16x16x32_bf16, fp32 accumulate).
// Graph ops (edge softmax, segment mean, weighted scatter) use f32 atomics
// that stay resident in the 192MB L2.
// ---------------------------------------------------------------------------

#define NH   4      // heads
#define DD   64     // per-head dim
#define HD   256    // NH*DD
#define MT   128    // GEMM block M tile
#define KT   32     // GEMM K step (WMMA K)
#define LSTRIDE 40  // LDS row stride in bf16 elems (32 + 8 pad)
#define CHUNK 65536 // rows per streamed edge-GEMM chunk

typedef __attribute__((ext_vector_type(16))) __bf16 v16bf;
typedef __attribute__((ext_vector_type(8)))  float  v8f;

union Frag { uint4 u[2]; v16bf v; };

__device__ __forceinline__ unsigned short f2bf(float f) {
  unsigned u = __float_as_uint(f);
  u += 0x7fffu + ((u >> 16) & 1u);          // round-to-nearest-even
  return (unsigned short)(u >> 16);
}
__device__ __forceinline__ unsigned sortable(float f) {
  unsigned u = __float_as_uint(f);
  return (u & 0x80000000u) ? ~u : (u | 0x80000000u);
}
__device__ __forceinline__ float unsortable(unsigned u) {
  return __uint_as_float((u & 0x80000000u) ? (u & 0x7fffffffu) : ~u);
}
__device__ __forceinline__ float leaky(float x) { return x >= 0.f ? x : 0.01f * x; }

// ---------------------------------------------------------------------------
// C[M,256] = (accum ? C : 0) + A[M,K](f32,row-major,lda) @ W[256,K](f32,ldw)^T + bias
// 8 waves/block, each wave: 16 rows x 256 cols = 16 C fragments.
// ---------------------------------------------------------------------------
__launch_bounds__(256)
__global__ void gemm_wmma(const float* __restrict__ A, int lda,
                          const float* __restrict__ W, int ldw,
                          const float* __restrict__ bias,
                          float* __restrict__ C,
                          int M, int K, int accum)
{
  __shared__ __align__(16) unsigned short As[MT * LSTRIDE];
  __shared__ __align__(16) unsigned short Ws[HD * LSTRIDE];
  const int tid = threadIdx.x, wave = tid >> 5, lane = tid & 31;
  const int m0 = blockIdx.x * MT;

  v8f acc[16];
#pragma unroll
  for (int i = 0; i < 16; ++i)
#pragma unroll
    for (int k = 0; k < 8; ++k) acc[i][k] = 0.f;

  for (int k0 = 0; k0 < K; k0 += KT) {
    // stage A tile: 128 x 32 f32 -> bf16 LDS (4 float4 per thread)
#pragma unroll
    for (int i = 0; i < 4; ++i) {
      int idx = tid + i * 256;
      int r = idx >> 3, c4 = (idx & 7) << 2;
      int row = m0 + r;
      float4 v = make_float4(0.f, 0.f, 0.f, 0.f);
      if (row < M) v = *(const float4*)(A + (size_t)row * lda + k0 + c4);
      unsigned short* p = &As[r * LSTRIDE + c4];
      p[0] = f2bf(v.x); p[1] = f2bf(v.y); p[2] = f2bf(v.z); p[3] = f2bf(v.w);
    }
    // stage W tile: 256 x 32 f32 -> bf16 LDS (8 float4 per thread)
#pragma unroll
    for (int i = 0; i < 8; ++i) {
      int idx = tid + i * 256;
      int r = idx >> 3, c4 = (idx & 7) << 2;
      float4 v = *(const float4*)(W + (size_t)r * ldw + k0 + c4);
      unsigned short* p = &Ws[r * LSTRIDE + c4];
      p[0] = f2bf(v.x); p[1] = f2bf(v.y); p[2] = f2bf(v.z); p[3] = f2bf(v.w);
    }
    __syncthreads();

    // A fragment: lane l<16 -> row, K {0..7,16..23}; lanes 16..31 -> +8
    Frag a;
    {
      int ar = wave * 16 + (lane & 15);
      int ak = (lane >> 4) << 3;
      a.u[0] = *(const uint4*)&As[ar * LSTRIDE + ak];
      a.u[1] = *(const uint4*)&As[ar * LSTRIDE + 16 + ak];
    }
#pragma unroll
    for (int nf = 0; nf < 16; ++nf) {
      // B fragment: lane l<16 -> col nf*16+l, K {0..7,16..23}; lanes 16..31 -> +8
      Frag b;
      int wr = nf * 16 + (lane & 15);
      int wk = (lane >> 4) << 3;
      b.u[0] = *(const uint4*)&Ws[wr * LSTRIDE + wk];
      b.u[1] = *(const uint4*)&Ws[wr * LSTRIDE + 16 + wk];
      acc[nf] = __builtin_amdgcn_wmma_f32_16x16x32_bf16(
          false, a.v, false, b.v, (short)0, acc[nf], false, false);
    }
    __syncthreads();
  }

  // epilogue: C frag layout -> lane l, vgpr r: row = r + 8*(l>>4), col = l&15
  const int rbase = m0 + wave * 16 + ((lane >> 4) << 3);
  const int c0 = lane & 15;
#pragma unroll
  for (int nf = 0; nf < 16; ++nf) {
    int col = nf * 16 + c0;
    float bv = bias ? bias[col] : 0.f;
#pragma unroll
    for (int r = 0; r < 8; ++r) {
      int row = rbase + r;
      if (row < M) {
        size_t idx = (size_t)row * HD + col;
        float v = acc[nf][r] + bv;
        if (accum) v += C[idx];
        C[idx] = v;
      }
    }
  }
}

// ---------------------------------------------------------------------------
// Per-edge: fo = leaky(xloc + g1[i1] + g2[i2] + bias); logits[h] = <fo_h, attn_h>
// optional fe[e][d] = sum_h fo[h*64+d]; atomicMax of logits into mbuf[i2].
// One wave per edge; lane handles cols {lane + 32j}.
// ---------------------------------------------------------------------------
__launch_bounds__(256)
__global__ void edge_attn(const float* __restrict__ xloc,
                          const float* __restrict__ g1,
                          const float* __restrict__ g2,
                          const int* __restrict__ i1,
                          const int* __restrict__ i2,
                          const float* __restrict__ bias,
                          const float* __restrict__ attn,
                          float* __restrict__ logits,
                          unsigned* __restrict__ mbuf,
                          float* __restrict__ fe,
                          long e0, int cnt)
{
  const int wave = threadIdx.x >> 5, lane = threadIdx.x & 31;
  const long e = e0 + (long)blockIdx.x * 8 + wave;
  if (e >= e0 + cnt) return;
  const int s = i1[e], d = i2[e];
  const float* xr = xloc + (size_t)(e - e0) * HD;
  const float* r1 = g1 + (size_t)s * HD;
  const float* r2 = g2 + (size_t)d * HD;
  float lh[NH] = {0.f, 0.f, 0.f, 0.f};
  float fe_lo = 0.f, fe_hi = 0.f;
#pragma unroll
  for (int j = 0; j < 8; ++j) {
    int c = lane + 32 * j;
    float v = leaky(xr[c] + r1[c] + r2[c] + bias[c]);
    lh[j >> 1] += v * attn[c];          // head = (lane+32j)/64 = j/2
    if (j & 1) fe_hi += v; else fe_lo += v;
  }
#pragma unroll
  for (int h = 0; h < NH; ++h)
#pragma unroll
    for (int off = 16; off > 0; off >>= 1) lh[h] += __shfl_xor(lh[h], off);
  if (fe) {
    fe[(size_t)e * DD + lane]      = fe_lo;
    fe[(size_t)e * DD + lane + 32] = fe_hi;
  }
  if (lane == 0) {
#pragma unroll
    for (int h = 0; h < NH; ++h) {
      logits[(size_t)e * NH + h] = lh[h];
      atomicMax(&mbuf[(size_t)d * NH + h], sortable(lh[h]));
    }
  }
}

// ------------------------- simple elementwise kernels -----------------------
__global__ void k_fill_f(float* p, float v, long n) {
  long i = (long)blockIdx.x * blockDim.x + threadIdx.x;
  if (i < n) p[i] = v;
}
__global__ void k_fill_u(unsigned* p, unsigned v, long n) {
  long i = (long)blockIdx.x * blockDim.x + threadIdx.x;
  if (i < n) p[i] = v;
}
__global__ void k_exp_sum(float* __restrict__ lg, const int* __restrict__ seg,
                          const unsigned* __restrict__ mbuf,
                          float* __restrict__ sbuf, long n) {
  long i = (long)blockIdx.x * blockDim.x + threadIdx.x;
  if (i >= n * NH) return;
  long e = i >> 2; int h = i & 3;
  int t = seg[e];
  float ex = __expf(lg[i] - unsortable(mbuf[(size_t)t * NH + h]));
  lg[i] = ex;
  atomicAdd(&sbuf[(size_t)t * NH + h], ex);
}
__global__ void k_norm(float* __restrict__ ex, const int* __restrict__ seg,
                       const float* __restrict__ s, long n) {
  long i = (long)blockIdx.x * blockDim.x + threadIdx.x;
  if (i >= n * NH) return;
  long e = i >> 2; int h = i & 3;
  ex[i] /= s[(size_t)seg[e] * NH + h];
}
__global__ void k_seg_accum(const float* __restrict__ feat, const int* __restrict__ seg,
                            float* __restrict__ acc, float* __restrict__ cnt,
                            long n, int shift) {
  long i = (long)blockIdx.x * blockDim.x + threadIdx.x;
  if (i >= (n << shift)) return;
  long e = i >> shift; int c = (int)(i & ((1 << shift) - 1));
  int t = seg[e];
  atomicAdd(&acc[((size_t)t << shift) + c], feat[i]);
  if (c == 0) atomicAdd(&cnt[t], 1.f);
}
__global__ void k_div_mean(float* acc, const float* __restrict__ cnt, long n, int shift) {
  long i = (long)blockIdx.x * blockDim.x + threadIdx.x;
  if (i >= (n << shift)) return;
  acc[i] /= fmaxf(cnt[i >> shift], 1.f);
}
// out[seg[e]] += msg[sidx[e]] * attn  (attn from edge row or from segment row)
__global__ void k_agg(const float* __restrict__ msg, const int* __restrict__ sidx,
                      const int* __restrict__ seg, const float* __restrict__ ex,
                      const float* __restrict__ sden, float* __restrict__ out,
                      long n, int attn_from_seg) {
  long i = (long)blockIdx.x * blockDim.x + threadIdx.x;
  if (i >= n * HD) return;
  long e = i >> 8; int c = (int)(i & 255); int h = c >> 6;
  int t = seg[e];
  float a = ex[(size_t)(attn_from_seg ? (long)t : e) * NH + h];
  if (sden) a /= sden[(size_t)t * NH + h];
  atomicAdd(&out[((size_t)t << 8) + c], msg[((size_t)sidx[e] << 8) + c] * a);
}
__global__ void k_node_edge(const float* __restrict__ hn, const int* __restrict__ src,
                            const int* __restrict__ dst, float* __restrict__ ne, long n) {
  long i = (long)blockIdx.x * blockDim.x + threadIdx.x;
  if (i >= n * DD) return;
  long e = i >> 6; int dd = (int)(i & 63);
  ne[i] = hn[(size_t)src[e] * DD + dd] + hn[(size_t)dst[e] * DD + dd];
}
// out[r][d] = (base ? base : 0) + sum_h leaky(acc[r][h*64+d])
__global__ void k_final(const float* __restrict__ acc, const float* __restrict__ base,
                        float* __restrict__ out, long n) {
  long i = (long)blockIdx.x * blockDim.x + threadIdx.x;
  if (i >= n * DD) return;
  long r = i >> 6; int dd = (int)(i & 63);
  const float* a = acc + ((size_t)r << 8) + dd;
  float v = leaky(a[0]) + leaky(a[64]) + leaky(a[128]) + leaky(a[192]);
  if (base) v += base[i];
  out[i] = v;
}

static inline long cdivl(long a, long b) { return (a + b - 1) / b; }

extern "C" void kernel_launch(void* const* d_in, const int* in_sizes, int n_in,
                              void* d_out, int out_size, void* d_ws, size_t ws_size,
                              hipStream_t stream)
{
  (void)n_in; (void)out_size; (void)ws_size;
  const float* nfeats  = (const float*)d_in[0];
  const float* efeats  = (const float*)d_in[1];
  const float* xfeats  = (const float*)d_in[2];
  const float* e_Wni   = (const float*)d_in[3];
  const float* e_Wnj   = (const float*)d_in[4];
  const float* e_Wfij  = (const float*)d_in[5];
  const float* e_Wnode = (const float*)d_in[6];
  const float* e_bnode = (const float*)d_in[7];
  const float* e_attn  = (const float*)d_in[8];
  const float* e_bias  = (const float*)d_in[9];
  const float* lg_Wni  = (const float*)d_in[10];
  const float* lg_Wnj  = (const float*)d_in[11];
  const float* lg_Wfij = (const float*)d_in[12];
  const float* lg_Wnode= (const float*)d_in[13];
  const float* lg_bnode= (const float*)d_in[14];
  const float* lg_attn = (const float*)d_in[15];
  const float* lg_bias = (const float*)d_in[16];
  const float* g_Wni   = (const float*)d_in[17];
  const float* g_Wnj   = (const float*)d_in[18];
  const float* g_Wfij  = (const float*)d_in[19];
  const float* g_Wnode = (const float*)d_in[20];
  const float* g_bnode = (const float*)d_in[21];
  const float* g_attn  = (const float*)d_in[22];
  const float* g_bias  = (const float*)d_in[23];
  const int* src  = (const int*)d_in[24];
  const int* dst  = (const int*)d_in[25];
  const int* lsrc = (const int*)d_in[26];
  const int* ldst = (const int*)d_in[27];

  const long N  = in_sizes[0] / 128;
  const long E  = in_sizes[1] / 128;
  const long EL = in_sizes[2] / 64;

  float* ws = (float*)d_ws;
  size_t off = 0;
  auto take = [&](size_t nelem) { float* p = ws + off; off += nelem; return p; };

  // persistent buffers
  float* f_edge = take((size_t)E * DD);
  float* h_node = take((size_t)N * DD);
  float* lg_h   = take((size_t)E * DD);
  const size_t SBASE = off;

  const int TB = 256;
  auto fillf = [&](float* p, float v, long n) {
    k_fill_f<<<(unsigned)cdivl(n, TB), TB, 0, stream>>>(p, v, n);
  };
  auto fillu = [&](unsigned* p, unsigned v, long n) {
    k_fill_u<<<(unsigned)cdivl(n, TB), TB, 0, stream>>>(p, v, n);
  };
  auto gemm = [&](const float* A, int lda, const float* W, int ldw, const float* bias,
                  float* C, long M, int K, int accum) {
    gemm_wmma<<<(unsigned)cdivl(M, MT), 256, 0, stream>>>(A, lda, W, ldw, bias, C,
                                                          (int)M, K, accum);
  };

  // =============== Phase A: EGAT on atom graph =============================
  {
    off = SBASE;
    float* Xn1  = take((size_t)N * HD);
    float* Xn2  = take((size_t)N * HD);
    float* chnk = take((size_t)CHUNK * HD);
    float* exA  = take((size_t)E * NH);
    unsigned* mA = (unsigned*)take((size_t)N * NH);
    float* sA   = take((size_t)N * NH);
    float* aggE = take((size_t)N * 128);
    float* cntA = take((size_t)N);
    float* Hn   = take((size_t)N * HD);
    float* hacc = take((size_t)N * HD);

    fillu(mA, 0u, N * NH);
    fillf(sA, 0.f, N * NH);
    fillf(aggE, 0.f, N * 128);
    fillf(cntA, 0.f, N);
    fillf(hacc, 0.f, N * HD);

    gemm(nfeats, 128, e_Wni, 128, nullptr, Xn1, N, 128, 0);
    gemm(nfeats, 128, e_Wnj, 128, nullptr, Xn2, N, 128, 0);
    for (long e0 = 0; e0 < E; e0 += CHUNK) {
      long cnt = (E - e0 < CHUNK) ? (E - e0) : CHUNK;
      gemm(efeats + e0 * 128, 128, e_Wfij, 128, nullptr, chnk, cnt, 128, 0);
      edge_attn<<<(unsigned)cdivl(cnt, 8), 256, 0, stream>>>(
          chnk, Xn1, Xn2, src, dst, e_bias, e_attn, exA, mA, f_edge, e0, (int)cnt);
    }
    k_exp_sum<<<(unsigned)cdivl(E * NH, TB), TB, 0, stream>>>(exA, dst, mA, sA, E);
    k_seg_accum<<<(unsigned)cdivl(E * 128, TB), TB, 0, stream>>>(efeats, dst, aggE, cntA, E, 7);
    k_div_mean<<<(unsigned)cdivl(N * 128, TB), TB, 0, stream>>>(aggE, cntA, N, 7);
    gemm(nfeats, 128, e_Wnode, 256, nullptr, Hn, N, 128, 0);
    gemm(aggE, 128, e_Wnode + 128, 256, e_bnode, Hn, N, 128, 1);
    k_agg<<<(unsigned)cdivl(E * HD, TB), TB, 0, stream>>>(Hn, src, dst, exA, sA, hacc, E, 0);
    k_final<<<(unsigned)cdivl(N * DD, TB), TB, 0, stream>>>(hacc, nullptr, h_node, N);
  }

  // =============== Phase B: line-graph (bond -> angle) branch ==============
  {
    off = SBASE;
    float* Y1   = take((size_t)E * HD);
    float* Y2   = take((size_t)E * HD);
    float* chnk = take((size_t)CHUNK * HD);
    float* exL  = take((size_t)EL * NH);
    unsigned* mL = (unsigned*)take((size_t)E * NH);
    float* sL   = take((size_t)E * NH);
    float* aggx = take((size_t)E * DD);
    float* cntX = take((size_t)E);
    float* M1   = take((size_t)E * HD);
    float* lacc = take((size_t)E * HD);

    fillu(mL, 0u, E * NH);
    fillf(sL, 0.f, E * NH);
    fillf(aggx, 0.f, E * DD);
    fillf(cntX, 0.f, E);
    fillf(lacc, 0.f, E * HD);

    gemm(f_edge, 64, lg_Wni, 64, nullptr, Y1, E, 64, 0);
    gemm(f_edge, 64, lg_Wnj, 64, nullptr, Y2, E, 64, 0);
    for (long e0 = 0; e0 < EL; e0 += CHUNK) {
      long cnt = (EL - e0 < CHUNK) ? (EL - e0) : CHUNK;
      gemm(xfeats + e0 * 64, 64, lg_Wfij, 64, nullptr, chnk, cnt, 64, 0);
      edge_attn<<<(unsigned)cdivl(cnt, 8), 256, 0, stream>>>(
          chnk, Y1, Y2, lsrc, ldst, lg_bias, lg_attn, exL, mL, nullptr, e0, (int)cnt);
    }
    k_exp_sum<<<(unsigned)cdivl(EL * NH, TB), TB, 0, stream>>>(exL, ldst, mL, sL, EL);
    k_seg_accum<<<(unsigned)cdivl(EL * DD, TB), TB, 0, stream>>>(xfeats, ldst, aggx, cntX, EL, 6);
    k_div_mean<<<(unsigned)cdivl(E * DD, TB), TB, 0, stream>>>(aggx, cntX, E, 6);
    gemm(f_edge, 64, lg_Wnode, 128, nullptr, M1, E, 64, 0);
    gemm(aggx, 64, lg_Wnode + 64, 128, lg_bnode, M1, E, 64, 1);
    k_agg<<<(unsigned)cdivl(EL * HD, TB), TB, 0, stream>>>(M1, lsrc, ldst, exL, sL, lacc, EL, 0);
    k_final<<<(unsigned)cdivl(E * DD, TB), TB, 0, stream>>>(lacc, nullptr, lg_h, E);
  }

  // =============== Phase C: graph (atom -> bond) branch ====================
  {
    off = SBASE;
    float* Z1   = take((size_t)N * HD);
    float* Z2   = take((size_t)N * HD);
    float* chnk = take((size_t)CHUNK * HD);
    float* aG   = take((size_t)E * NH);
    unsigned* mG = (unsigned*)take((size_t)N * NH);
    float* sG   = take((size_t)N * NH);
    float* ne   = take((size_t)E * DD);
    float* M2   = take((size_t)E * HD);
    float* gacc = take((size_t)E * HD);

    fillu(mG, 0u, N * NH);
    fillf(sG, 0.f, N * NH);
    fillf(gacc, 0.f, E * HD);

    gemm(h_node, 64, g_Wni, 64, nullptr, Z1, N, 64, 0);
    gemm(h_node, 64, g_Wnj, 64, nullptr, Z2, N, 64, 0);
    for (long e0 = 0; e0 < E; e0 += CHUNK) {
      long cnt = (E - e0 < CHUNK) ? (E - e0) : CHUNK;
      gemm(f_edge + e0 * 64, 64, g_Wfij, 64, nullptr, chnk, cnt, 64, 0);
      edge_attn<<<(unsigned)cdivl(cnt, 8), 256, 0, stream>>>(
          chnk, Z1, Z2, src, dst, g_bias, g_attn, aG, mG, nullptr, e0, (int)cnt);
    }
    k_exp_sum<<<(unsigned)cdivl(E * NH, TB), TB, 0, stream>>>(aG, dst, mG, sG, E);
    k_norm<<<(unsigned)cdivl(E * NH, TB), TB, 0, stream>>>(aG, dst, sG, E);
    k_node_edge<<<(unsigned)cdivl(E * DD, TB), TB, 0, stream>>>(h_node, src, dst, ne, E);
    gemm(ne, 64, g_Wnode, 128, nullptr, M2, E, 64, 0);
    gemm(f_edge, 64, g_Wnode + 64, 128, g_bnode, M2, E, 64, 1);
    k_agg<<<(unsigned)cdivl(EL * HD, TB), TB, 0, stream>>>(M2, lsrc, ldst, aG, nullptr, gacc, EL, 1);
    k_final<<<(unsigned)cdivl(E * DD, TB), TB, 0, stream>>>(gacc, lg_h, (float*)d_out, E);
  }
}
